// Position_attention_15951508537564
// MI455X (gfx1250) — matrix operand order repaired and checked by reading, not compiled, not run on hardware
//
#include <hip/hip_runtime.h>
#include <hip/hip_bf16.h>

// ---------------------------------------------------------------------------
// Position (channel) attention, fused for MI455X / gfx1250, wave32 + WMMA bf16
//   B=4, D=16, H=64, W=64, C=128  ->  N = 65536 positions per batch
//
// Roofline: ~43 GFLOP total vs ~400 MB min HBM traffic when fully fused
// (x read twice, out written once, 128x128 energy/attn kept in L2).
// => memory-bound at 23.3 TB/s (~17 us); bf16 WMMA keeps compute far ahead.
// ---------------------------------------------------------------------------

typedef __bf16 v16bf __attribute__((ext_vector_type(16)));
typedef __bf16 v8bf  __attribute__((ext_vector_type(8)));
typedef __bf16 v4bf  __attribute__((ext_vector_type(4)));
typedef float  v8f   __attribute__((ext_vector_type(8)));
typedef float  v4f   __attribute__((ext_vector_type(4)));

#define CCH   128            // channels
#define NPOS  65536          // positions per batch (16*64*64)
#define TPOS  32             // positions per WMMA step (K of energy GEMM)
#define CHB   64             // chunks per batch
#define CHUNK (NPOS / CHB)   // 1024 positions per workgroup
#define NB    4              // batches

// ---- WMMA wrapper ----------------------------------------------------------
__device__ inline v8f wmma_bf16(v16bf a, v16bf b, v8f c) {
    // D = A(16x32 bf16) * B(32x16 bf16) + C(16x16 f32)
    return __builtin_amdgcn_wmma_f32_16x16x32_bf16(
        /*neg_a=*/false, a, /*neg_b=*/false, b,
        /*c_mod=*/(short)0, c, /*reuse_a=*/false, /*reuse_b=*/false);
}

// A fragment (M=16 x K=32, bf16). base -> element [m=0][k=0], ld = row stride
// (elements). Lane l: m = l&15, half = l>>4; reads K = 16*half..+15
// => one 32-byte contiguous chunk per lane.
__device__ inline v16bf loadA(const __bf16* base, int ld, int lane) {
    const int m = lane & 15, half = lane >> 4;
    const __bf16* row = base + m * ld + 16 * half;
    return *(const v16bf*)(row);
}

// B fragment (K=32 x N=16, bf16) from a K-contiguous layout: base -> element
// [k=0][n=0] of an N-major array (row n holds K contiguous). Lane l: n = l&15,
// reads K = 16*(l>>4) .. +15 => one 32-byte contiguous chunk.
__device__ inline v16bf loadB(const __bf16* base, int ld, int lane) {
    const int n = lane & 15, kb = (lane >> 4) * 16;
    return *(const v16bf*)(base + n * ld + kb);
}

// Vectorized f32 -> bf16 tile staging: b128 global loads, 8-byte LDS stores.
__device__ inline void stage_tile_bf16(const float* __restrict__ src,
                                       __bf16* __restrict__ dst,
                                       int nelem, int tid, int nthr) {
    const v4f* s4 = (const v4f*)src;
    for (int i = tid; i < nelem / 4; i += nthr) {
        v4f f = s4[i];
        v4bf h;
        h[0] = (__bf16)f[0]; h[1] = (__bf16)f[1];
        h[2] = (__bf16)f[2]; h[3] = (__bf16)f[3];
        *(v4bf*)&dst[i * 4] = h;
    }
}

// ---- kernel 0: zero the energy accumulator --------------------------------
__global__ void pa_zero_f32(float* __restrict__ p, int n) {
    int i = blockIdx.x * blockDim.x + threadIdx.x;
    if (i < n) p[i] = 0.0f;
}

// ---- kernel 1: fused q/k projection + energy = q^T k ----------------------
// grid = NB*CHB workgroups of 256 threads (8 waves). Wave w owns energy rows
// (q-channels) [16w,16w+16) x all 128 k-channels, accumulated in 8 v8f tiles.
__launch_bounds__(256, 1)
__global__ void pa_energy_kernel(const float* __restrict__ x,
                                 const float* __restrict__ Wq,
                                 const float* __restrict__ bq,
                                 const float* __restrict__ Wk,
                                 const float* __restrict__ bk,
                                 float* __restrict__ energy) {
    __shared__ __align__(32) __bf16 sWq[CCH * CCH];   // [out][in] (transposed)
    __shared__ __align__(32) __bf16 sWk[CCH * CCH];   // [out][in]
    __shared__ __align__(32) __bf16 sX [TPOS * CCH];  // pos-major  [pos][in]
    __shared__ __align__(32) __bf16 sQ [CCH * TPOS];  // chan-major [chan][pos]
    __shared__ __align__(32) __bf16 sK [CCH * TPOS];  // chan-major [chan][pos]
    __shared__ float sBq[CCH], sBk[CCH];

    const int tid  = threadIdx.x;
    const int wave = tid >> 5;
    const int lane = tid & 31;
    const int b    = blockIdx.x / CHB;
    const int chnk = blockIdx.x % CHB;
    const float* xb = x + (size_t)b * NPOS * CCH + (size_t)chnk * CHUNK * CCH;

    // Stage weights transposed (W[in][out] -> sW[out*C+in]) and biases.
    // b128 global loads (4 contiguous "out"), scalar b16 transposed stores.
    {
        const v4f* wq4 = (const v4f*)Wq;
        const v4f* wk4 = (const v4f*)Wk;
        for (int i = tid; i < CCH * CCH / 4; i += 256) {
            const int in = (i * 4) / CCH, o0 = (i * 4) % CCH;
            v4f fq = wq4[i], fk = wk4[i];
            for (int j = 0; j < 4; ++j) {
                sWq[(o0 + j) * CCH + in] = (__bf16)fq[j];
                sWk[(o0 + j) * CCH + in] = (__bf16)fk[j];
            }
        }
    }
    for (int i = tid; i < CCH; i += 256) { sBq[i] = bq[i]; sBk[i] = bk[i]; }
    __syncthreads();

    v8f accE[8];
    for (int t = 0; t < 8; ++t)
        for (int e = 0; e < 8; ++e) accE[t][e] = 0.0f;

    const int dchan = wave * 16 + (lane & 15);   // D-tile channel of this lane
    const int half  = lane >> 4;

    for (int it = 0; it < CHUNK / TPOS; ++it) {
        const float* xt = xb + (size_t)it * TPOS * CCH;
        __builtin_prefetch(xt + TPOS * CCH, 0, 1);   // global_prefetch next tile

        stage_tile_bf16(xt, sX, TPOS * CCH, tid, 256);
        __syncthreads();

        // q,k tiles: wave w -> out-channel tile w, pos tiles 0..1, K = 128.
        for (int pt = 0; pt < 2; ++pt) {
            v8f aq, ak;
            for (int e = 0; e < 8; ++e) { aq[e] = 0.0f; ak[e] = 0.0f; }
            for (int ks = 0; ks < CCH / 32; ++ks) {
                v16bf a  = loadA(&sX [(pt * 16) * CCH + ks * 32], CCH, lane);
                v16bf wq = loadB(&sWq[(wave * 16) * CCH + ks * 32], CCH, lane);
                v16bf wk = loadB(&sWk[(wave * 16) * CCH + ks * 32], CCH, lane);
                aq = wmma_bf16(a, wq, aq);
                ak = wmma_bf16(a, wk, ak);
            }
            // D layout: pos = pt*16 + r + 8*half, chan = dchan. Store chan-major
            // => 8 consecutive bf16 per lane, one b128 LDS store.
            const int pos0 = pt * 16 + 8 * half;
            v8bf vq, vk;
            for (int r = 0; r < 8; ++r) {
                vq[r] = (__bf16)(aq[r] + sBq[dchan]);
                vk[r] = (__bf16)(ak[r] + sBk[dchan]);
            }
            *(v8bf*)&sQ[dchan * TPOS + pos0] = vq;
            *(v8bf*)&sK[dchan * TPOS + pos0] = vk;
        }
        __syncthreads();

        // energy += q^T(16x32) * k(32x16) for 8 k-channel tiles.
        v16bf aE = loadA(&sQ[(wave * 16) * TPOS], TPOS, lane);
        for (int nt = 0; nt < 8; ++nt) {
            v16bf bE = loadB(&sK[(nt * 16) * TPOS], TPOS, lane);
            accE[nt] = wmma_bf16(aE, bE, accE[nt]);
        }
        __syncthreads();
    }

    // Reduce partial energies across workgroups: f32 global atomics into the
    // tiny (L2-resident) 128x128 per-batch accumulator.
    float* eb = energy + (size_t)b * CCH * CCH;
    for (int nt = 0; nt < 8; ++nt) {
        const int n = nt * 16 + (lane & 15);
        for (int r = 0; r < 8; ++r) {
            const int m = wave * 16 + r + 8 * half;
            __hip_atomic_fetch_add(&eb[m * CCH + n], accE[nt][r],
                                   __ATOMIC_RELAXED, __HIP_MEMORY_SCOPE_AGENT);
        }
    }
}

// ---- kernel 2: softmax over last axis of (B, C, C) ------------------------
__launch_bounds__(128, 4)
__global__ void pa_softmax_kernel(const float* __restrict__ energy,
                                  float* __restrict__ attn) {
    __shared__ float red[CCH];
    const int row = blockIdx.x;        // B*C rows
    const int t   = threadIdx.x;       // 0..127
    const float v = energy[(size_t)row * CCH + t];

    red[t] = v;
    __syncthreads();
    for (int s = 64; s > 0; s >>= 1) {
        if (t < s) red[t] = fmaxf(red[t], red[t + s]);
        __syncthreads();
    }
    const float m = red[0];
    __syncthreads();
    const float e = __expf(v - m);
    red[t] = e;
    __syncthreads();
    for (int s = 64; s > 0; s >>= 1) {
        if (t < s) red[t] += red[t + s];
        __syncthreads();
    }
    attn[(size_t)row * CCH + t] = e / red[0];
}

// ---- kernel 3: fused v projection + out = v*attn^T, gamma*out + x ---------
__launch_bounds__(256, 1)
__global__ void pa_out_kernel(const float* __restrict__ x,
                              const float* __restrict__ Wv,
                              const float* __restrict__ bv,
                              const float* __restrict__ attn,
                              const float* __restrict__ gamma,
                              float* __restrict__ out) {
    __shared__ __align__(32) __bf16 sWv[CCH * CCH];   // [out][in] (transposed)
    __shared__ __align__(32) __bf16 sA [CCH * CCH];   // attn row-major [c'][c]
    __shared__ __align__(32) __bf16 sX [TPOS * CCH];  // pos-major
    __shared__ __align__(32) __bf16 sV [TPOS * CCH];  // pos-major
    __shared__ float sBv[CCH], sG[CCH];

    const int tid  = threadIdx.x;
    const int wave = tid >> 5;
    const int lane = tid & 31;
    const int half = lane >> 4;
    const int b    = blockIdx.x / CHB;
    const int chnk = blockIdx.x % CHB;
    const float* xb = x    + (size_t)b * NPOS * CCH + (size_t)chnk * CHUNK * CCH;
    float*       ob = out  + (size_t)b * NPOS * CCH + (size_t)chnk * CHUNK * CCH;
    const float* ab = attn + (size_t)b * CCH * CCH;

    {
        const v4f* wv4 = (const v4f*)Wv;
        for (int i = tid; i < CCH * CCH / 4; i += 256) {
            const int in = (i * 4) / CCH, o0 = (i * 4) % CCH;
            v4f fv = wv4[i];
            for (int j = 0; j < 4; ++j)
                sWv[(o0 + j) * CCH + in] = (__bf16)fv[j];
        }
    }
    stage_tile_bf16(ab, sA, CCH * CCH, tid, 256);  // B frag: lane=c', contiguous c
    for (int i = tid; i < CCH; i += 256) { sBv[i] = bv[i]; sG[i] = gamma[i]; }
    __syncthreads();

    const int dchan = wave * 16 + (lane & 15);

    for (int it = 0; it < CHUNK / TPOS; ++it) {
        const float* xt = xb + (size_t)it * TPOS * CCH;
        float*       ot = ob + (size_t)it * TPOS * CCH;
        __builtin_prefetch(xt + TPOS * CCH, 0, 1);

        stage_tile_bf16(xt, sX, TPOS * CCH, tid, 256);
        __syncthreads();

        // v tile (32x128): wave w -> channel tile w, pos tiles 0..1.
        for (int pt = 0; pt < 2; ++pt) {
            v8f av;
            for (int e = 0; e < 8; ++e) av[e] = 0.0f;
            for (int ks = 0; ks < CCH / 32; ++ks) {
                v16bf a  = loadA(&sX [(pt * 16) * CCH + ks * 32], CCH, lane);
                v16bf wv = loadB(&sWv[(wave * 16) * CCH + ks * 32], CCH, lane);
                av = wmma_bf16(a, wv, av);
            }
            for (int r = 0; r < 8; ++r) {
                const int pos = pt * 16 + r + 8 * half;
                sV[pos * CCH + dchan] = (__bf16)(av[r] + sBv[dchan]);
            }
        }
        __syncthreads();

        // out tile: wave w -> pos tile (w>>2), c'-tiles 2*(w&3), 2*(w&3)+1.
        const int pt = wave >> 2;
        for (int cc = 0; cc < 2; ++cc) {
            const int ct = (wave & 3) * 2 + cc;
            v8f ao;
            for (int e = 0; e < 8; ++e) ao[e] = 0.0f;
            for (int ks = 0; ks < CCH / 32; ++ks) {
                v16bf a  = loadA(&sV[(pt * 16) * CCH + ks * 32], CCH, lane);
                v16bf bm = loadB(&sA[(ct * 16) * CCH + ks * 32], CCH, lane);
                ao = wmma_bf16(a, bm, ao);
            }
            const int chan = ct * 16 + (lane & 15);
            const float g = sG[chan];
            for (int r = 0; r < 8; ++r) {
                const int pos = pt * 16 + r + 8 * half;
                // exact f32 residual from global (coalesced across half-wave)
                ot[pos * CCH + chan] = g * ao[r] + xt[pos * CCH + chan];
            }
        }
        __syncthreads();
    }
}

// ---------------------------------------------------------------------------
extern "C" void kernel_launch(void* const* d_in, const int* in_sizes, int n_in,
                              void* d_out, int out_size, void* d_ws, size_t ws_size,
                              hipStream_t stream) {
    const float* x     = (const float*)d_in[0];
    const float* Wq    = (const float*)d_in[1];
    const float* bq    = (const float*)d_in[2];
    const float* Wk    = (const float*)d_in[3];
    const float* bk    = (const float*)d_in[4];
    const float* Wv    = (const float*)d_in[5];
    const float* bv    = (const float*)d_in[6];
    const float* gamma = (const float*)d_in[7];
    float* out = (float*)d_out;

    float* energy = (float*)d_ws;                 // NB*C*C f32 = 256 KB
    float* attn   = energy + (size_t)NB * CCH * CCH;

    const int ne = NB * CCH * CCH;
    pa_zero_f32<<<(ne + 255) / 256, 256, 0, stream>>>(energy, ne);
    pa_energy_kernel<<<NB * CHB, 256, 0, stream>>>(x, Wq, bq, Wk, bk, energy);
    pa_softmax_kernel<<<NB * CCH, 128, 0, stream>>>(energy, attn);
    pa_out_kernel<<<NB * CHB, 256, 0, stream>>>(x, Wv, bv, attn, gamma, out);
}
